// GNNPolicyNetwork_25744033972726
// MI455X (gfx1250) — matrix operand order repaired and checked by reading, not compiled, not run on hardware
//
#include <hip/hip_runtime.h>
#include <hip/hip_bf16.h>
#include <math.h>

typedef __attribute__((ext_vector_type(16))) _Float16 v16h;
typedef __attribute__((ext_vector_type(8)))  float    v8f;
typedef int v4i_ __attribute__((vector_size(16)));   // matches builtin param type

#define N_NODES   100000
#define N_EDGES   1600000
#define DIM       128
#define N_GRAPHS  64
#define D_OUT     32
#define GEMM_ROWS 64
#define ROWS_PAD  100032   // N_NODES padded to multiple of GEMM_ROWS
#define XLD       136      // padded LDS stride for the (tiny) head kernel

#if __has_builtin(__builtin_amdgcn_global_load_async_to_lds_b128) && \
    __has_builtin(__builtin_amdgcn_s_wait_asynccnt)
#define HAVE_ASYNC_LDS 1
#endif

// ---------------------------------------------------------------------------
// WMMA fragment helpers for the small head kernel (layouts per ISA 7.12.2)
// ---------------------------------------------------------------------------
__device__ inline v16h a_frag_lds(const _Float16* s, int ld, int row0, int k0, int lane) {
  int m    = row0 + (lane & 15);
  int half = lane >> 4;
  v16h a;
#pragma unroll
  for (int i = 0; i < 16; ++i) {
    int k = k0 + ((i & 8) ? 16 : 0) + half * 8 + (((i >> 1) & 3) << 1) + (i & 1);
    a[i] = s[m * ld + k];
  }
  return a;
}

__device__ inline v16h b_frag_gmem_f32(const float* W, int ld, int k0, int col0, int lane) {
  int n    = col0 + (lane & 15);
  int half = lane >> 4;
  v16h b;
#pragma unroll
  for (int i = 0; i < 16; ++i) {
    int k = k0 + ((i & 8) ? 16 : 0) + half * 8 + (i & 7);
    b[i] = (_Float16)W[k * ld + n];
  }
  return b;
}

// ---------------------------------------------------------------------------
// Utility kernels
// ---------------------------------------------------------------------------
__global__ void k_zero(float* p, long n) {
  long i = (long)blockIdx.x * blockDim.x + threadIdx.x;
  long stride = (long)gridDim.x * blockDim.x;
  for (; i < n; i += stride) p[i] = 0.0f;
}

__global__ void k_zero_h(_Float16* p, long n) {
  long i = (long)blockIdx.x * blockDim.x + threadIdx.x;
  long stride = (long)gridDim.x * blockDim.x;
  for (; i < n; i += stride) p[i] = (_Float16)0.0f;
}

__global__ void k_degree(const int* __restrict__ dst, float* __restrict__ deg, int E) {
  int e = blockIdx.x * blockDim.x + threadIdx.x;
  if (e < E) atomicAdd(&deg[dst[e]], 1.0f);
}

__global__ void k_dinv(float* d, int n) {
  int i = blockIdx.x * blockDim.x + threadIdx.x;
  if (i < n) d[i] = rsqrtf(1.0f + d[i]);
}

// x (f32, N rows) -> xh (f16, ROWS_PAD rows, zero-padded tail), row-major
__global__ void k_cvt_x(const float* __restrict__ x, _Float16* __restrict__ xh, int n_rows) {
  long t = (long)blockIdx.x * blockDim.x + threadIdx.x;
  if (t >= (long)ROWS_PAD * DIM) return;
  int row = (int)(t >> 7);
  xh[t] = (row < n_rows) ? (_Float16)x[t] : (_Float16)0.0f;
}

// W[128x128] f32 row-major -> fragment-ordered f16:
// frag f = ct*4+kk holds B-tile (cols ct*16.., k kk*32..); lane's 16 halves contiguous.
__global__ void k_prep_w(const float* __restrict__ W, _Float16* __restrict__ Wsw) {
  int idx = blockIdx.x * blockDim.x + threadIdx.x;
  if (idx >= DIM * DIM) return;
  int k = idx >> 7, n = idx & 127;
  int ct = n >> 4, kk = k >> 5, kl = k & 31;
  int lane = (n & 15) + (((kl >> 3) & 1) << 4);
  int i = (kl & 7) + ((kl >> 4) << 3);
  Wsw[((ct * 4 + kk) * 32 + lane) * 16 + i] = (_Float16)W[idx];
}

// ---------------------------------------------------------------------------
// Dense GEMM: H[ROWS_PAD x 128] = Xh @ W, f16 in / f32 accumulate.
// 256 threads (8 waves), 64 rows per block. X tile staged fragment-ordered in
// LDS via async-to-LDS B128 (per-lane global + LDS addresses); B fragments
// loaded as coalesced 32B vectors from pre-swizzled global W (L2-resident).
// ---------------------------------------------------------------------------
__global__ __launch_bounds__(256) void k_gemm_node(const _Float16* __restrict__ Xh,
                                                   const _Float16* __restrict__ Wsw,
                                                   float* __restrict__ H) {
  __shared__ v16h sX[16 * 32];  // 16 fragments x 32 lanes x 32B = 16 KB
  const int tid  = threadIdx.x;
  const int wave = tid >> 5;
  const int lane = tid & 31;
  const int half = lane >> 4;
  const int m    = lane & 15;
  const long row_base = (long)blockIdx.x * GEMM_ROWS;

  // Stage X tile: wave w fills fragments f = 2w, 2w+1 (f = rt*4 + kk).
  // Each lane moves two contiguous 16B chunks: k_local = half*8+[0,8) and 16+half*8+[0,8).
#pragma unroll
  for (int j = 0; j < 2; ++j) {
    int f  = wave * 2 + j;
    int rt = f >> 2, kk = f & 3;
    long grow = row_base + rt * 16 + m;
    const _Float16* gsrc = Xh + grow * DIM + kk * 32 + half * 8;
    _Float16* ldst = (_Float16*)&sX[f * 32 + lane];
#if defined(HAVE_ASYNC_LDS)
    __builtin_amdgcn_global_load_async_to_lds_b128(
        (__attribute__((address_space(1))) v4i_*)gsrc,
        (__attribute__((address_space(3))) v4i_*)ldst, 0, 0);
    __builtin_amdgcn_global_load_async_to_lds_b128(
        (__attribute__((address_space(1))) v4i_*)(gsrc + 16),
        (__attribute__((address_space(3))) v4i_*)(ldst + 8), 0, 0);
#else
    *(uint4*)ldst       = *(const uint4*)gsrc;
    *(uint4*)(ldst + 8) = *(const uint4*)(gsrc + 16);
#endif
  }
#if defined(HAVE_ASYNC_LDS)
  __builtin_amdgcn_s_wait_asynccnt(0);
#endif
  __syncthreads();

  // B fragments for this wave's column tile: single 32B vector load each.
  v16h bf[4];
#pragma unroll
  for (int kk = 0; kk < 4; ++kk)
    bf[kk] = *(const v16h*)(Wsw + ((wave * 4 + kk) * 32 + lane) * 16);

  const int col = wave * 16 + m;
#pragma unroll
  for (int rt = 0; rt < 4; ++rt) {
    v8f acc = {};
#pragma unroll
    for (int kk = 0; kk < 4; ++kk)
      acc = __builtin_amdgcn_wmma_f32_16x16x32_f16(false, sX[(rt * 4 + kk) * 32 + lane],
                                                   false, bf[kk], (short)0, acc,
                                                   false, false);
    // Unguarded stores: H is padded to ROWS_PAD rows.
#pragma unroll
    for (int r = 0; r < 8; ++r) {
      long grow = row_base + rt * 16 + r + half * 8;
      H[grow * DIM + col] = acc[r];
    }
  }
}

// ---------------------------------------------------------------------------
// Edge aggregation: agg[dst] += h[src] * dinv[src]*dinv[dst]; 1 wave/edge,
// 4 cols/lane (float4 gather + 4 f32 atomics).
// ---------------------------------------------------------------------------
__global__ void k_aggregate(const float* __restrict__ H, const int* __restrict__ src,
                            const int* __restrict__ dst, const float* __restrict__ dinv,
                            float* __restrict__ agg, int E) {
  long t = (long)blockIdx.x * blockDim.x + threadIdx.x;
  int e = (int)(t >> 5);
  if (e >= E) return;
  int lane = (int)(t & 31);
  int s = src[e], d = dst[e];
  float norm = dinv[s] * dinv[d];
  int c = lane * 4;
  const float4 v = *(const float4*)(H + (long)s * DIM + c);
  float* ap = agg + (long)d * DIM + c;
  atomicAdd(ap + 0, v.x * norm);
  atomicAdd(ap + 1, v.y * norm);
  atomicAdd(ap + 2, v.z * norm);
  atomicAdd(ap + 3, v.w * norm);
}

// h = relu(agg + h*dinv^2 + bias); writes f32 in place AND f16 copy for next GEMM
__global__ void k_finalize(float* __restrict__ h, _Float16* __restrict__ h16,
                           const float* __restrict__ agg, const float* __restrict__ dinv,
                           const float* __restrict__ bias, int n) {
  long t = (long)blockIdx.x * blockDim.x + threadIdx.x;
  if (t >= (long)n * DIM) return;
  int i = (int)(t >> 7);
  int c = (int)(t & 127);
  float di = dinv[i];
  float v = agg[t] + h[t] * di * di + bias[c];
  v = v > 0.0f ? v : 0.0f;
  h[t]   = v;
  h16[t] = (_Float16)v;
}

// Mean-pool accumulation: 1 wave per node, 4 cols per lane
__global__ void k_pool(const float* __restrict__ h, const int* __restrict__ batch,
                       float* __restrict__ sums, float* __restrict__ cnts, int n) {
  long t = (long)blockIdx.x * blockDim.x + threadIdx.x;
  int node = (int)(t >> 5);
  if (node >= n) return;
  int lane = (int)(t & 31);
  int g = batch[node];
  int c = lane * 4;
  const float4 v = *(const float4*)(h + (long)node * DIM + c);
  float* sp = sums + g * DIM + c;
  atomicAdd(sp + 0, v.x);
  atomicAdd(sp + 1, v.y);
  atomicAdd(sp + 2, v.z);
  atomicAdd(sp + 3, v.w);
  if (lane == 0) atomicAdd(&cnts[g], 1.0f);
}

__global__ void k_pool_div(float* sums, const float* cnts) {
  int t = blockIdx.x * blockDim.x + threadIdx.x;
  if (t < N_GRAPHS * DIM) sums[t] /= fmaxf(cnts[t >> 7], 1.0f);
}

// ---------------------------------------------------------------------------
// MLP head + softmax, single block of 256 threads (8 waves), WMMA throughout.
// ---------------------------------------------------------------------------
__global__ __launch_bounds__(256) void k_head(const float* __restrict__ g,
                                              const float* __restrict__ fW1,
                                              const float* __restrict__ fb1,
                                              const float* __restrict__ fW2,
                                              const float* __restrict__ fb2,
                                              float* __restrict__ out) {
  __shared__ _Float16 sg[N_GRAPHS * XLD];
  __shared__ _Float16 sg1[N_GRAPHS * XLD];
  __shared__ float    slog[N_GRAPHS * D_OUT];
  int tid = threadIdx.x, lane = tid & 31, wave = tid >> 5;
  int half = lane >> 4;

#pragma unroll
  for (int i = 0; i < (N_GRAPHS * DIM) / 256; ++i) {
    int idx = tid + i * 256;
    int r = idx >> 7, c = idx & 127;
    sg[r * XLD + c] = (_Float16)g[idx];
  }
  __syncthreads();

  // GEMM1: 64x128 @ 128x128, +bias, relu -> sg1 (f16)
  {
    v16h bf[4];
#pragma unroll
    for (int kk = 0; kk < 4; ++kk)
      bf[kk] = b_frag_gmem_f32(fW1, DIM, kk * 32, wave * 16, lane);
    int col = wave * 16 + (lane & 15);
#pragma unroll
    for (int rt = 0; rt < 4; ++rt) {
      v8f acc = {};
#pragma unroll
      for (int kk = 0; kk < 4; ++kk) {
        v16h a = a_frag_lds(sg, XLD, rt * 16, kk * 32, lane);
        acc = __builtin_amdgcn_wmma_f32_16x16x32_f16(false, a, false, bf[kk],
                                                     (short)0, acc, false, false);
      }
#pragma unroll
      for (int r = 0; r < 8; ++r) {
        int row = rt * 16 + r + half * 8;
        float v = acc[r] + fb1[col];
        sg1[row * XLD + col] = (_Float16)(v > 0.0f ? v : 0.0f);
      }
    }
  }
  __syncthreads();

  // GEMM2: 64x128 @ 128x32 -> logits in LDS
  {
    int rt = wave >> 1, ct = wave & 1;
    int col = ct * 16 + (lane & 15);
    v8f acc = {};
#pragma unroll
    for (int kk = 0; kk < 4; ++kk) {
      v16h a = a_frag_lds(sg1, XLD, rt * 16, kk * 32, lane);
      v16h b = b_frag_gmem_f32(fW2, D_OUT, kk * 32, ct * 16, lane);
      acc = __builtin_amdgcn_wmma_f32_16x16x32_f16(false, a, false, b,
                                                   (short)0, acc, false, false);
    }
#pragma unroll
    for (int r = 0; r < 8; ++r) {
      int row = rt * 16 + r + half * 8;
      slog[row * D_OUT + col] = acc[r] + fb2[col];
    }
  }
  __syncthreads();

  // Softmax: one 32-col row per wave, wave32 shuffle reductions
  for (int it = 0; it < 8; ++it) {
    int row = wave + it * 8;
    float v = slog[row * D_OUT + lane];
    float mx = v;
#pragma unroll
    for (int off = 16; off >= 1; off >>= 1) mx = fmaxf(mx, __shfl_xor(mx, off, 32));
    float e = expf(v - mx);
    float s = e;
#pragma unroll
    for (int off = 16; off >= 1; off >>= 1) s += __shfl_xor(s, off, 32);
    out[row * D_OUT + lane] = e / s;
  }
}

// ---------------------------------------------------------------------------
extern "C" void kernel_launch(void* const* d_in, const int* in_sizes, int n_in,
                              void* d_out, int out_size, void* d_ws, size_t ws_size,
                              hipStream_t stream) {
  (void)in_sizes; (void)n_in; (void)out_size; (void)ws_size;
  const float* x          = (const float*)d_in[0];
  const int*   edge_index = (const int*)d_in[1];
  const int*   batch      = (const int*)d_in[2];
  const float* W1  = (const float*)d_in[3];
  const float* b1  = (const float*)d_in[4];
  const float* W2  = (const float*)d_in[5];
  const float* b2  = (const float*)d_in[6];
  const float* fW1 = (const float*)d_in[7];
  const float* fb1 = (const float*)d_in[8];
  const float* fW2 = (const float*)d_in[9];
  const float* fb2 = (const float*)d_in[10];
  float* out = (float*)d_out;

  const int* src = edge_index;
  const int* dst = edge_index + N_EDGES;

  // Workspace layout (float units):
  // dinv | buf1 (padded) | buf2 (padded) | xh (f16) | h16 (f16) | w1sw | w2sw | sums | cnts
  const long PH = (long)ROWS_PAD * DIM;  // 12,804,096
  float* ws   = (float*)d_ws;
  float* dinv = ws;
  float* buf1 = ws + 100096;
  float* buf2 = buf1 + PH;
  _Float16* xh  = (_Float16*)(buf2 + PH);
  _Float16* h16 = xh + PH;
  _Float16* w1s = h16 + PH;
  _Float16* w2s = w1s + DIM * DIM;
  float* sums = (float*)(w2s + DIM * DIM);
  float* cnts = sums + N_GRAPHS * DIM;

  const int T = 256;
  const long NH = (long)N_NODES * DIM;
  const int gemm_blocks = ROWS_PAD / GEMM_ROWS;                       // 1563
  const int agg_blocks  = (int)(((long)N_EDGES * 32 + T - 1) / T);
  const int fin_blocks  = (int)((NH + T - 1) / T);

  // degree / dinv
  k_zero<<<1024, T, 0, stream>>>(dinv, N_NODES);
  k_degree<<<(N_EDGES + T - 1) / T, T, 0, stream>>>(dst, dinv, N_EDGES);
  k_dinv<<<(N_NODES + T - 1) / T, T, 0, stream>>>(dinv, N_NODES);

  // One-time prep: x -> f16 padded; W1/W2 -> fragment-ordered f16; zero h16 pad tail
  k_cvt_x<<<(int)((PH + T - 1) / T), T, 0, stream>>>(x, xh, N_NODES);
  k_prep_w<<<(DIM * DIM + T - 1) / T, T, 0, stream>>>(W1, w1s);
  k_prep_w<<<(DIM * DIM + T - 1) / T, T, 0, stream>>>(W2, w2s);
  k_zero_h<<<16, T, 0, stream>>>(h16 + NH, PH - NH);

  // Layer 1 -> buf1 (f32) + h16 (f16)
  k_gemm_node<<<gemm_blocks, T, 0, stream>>>(xh, w1s, buf1);
  k_zero<<<4096, T, 0, stream>>>(buf2, NH);
  k_aggregate<<<agg_blocks, T, 0, stream>>>(buf1, src, dst, dinv, buf2, N_EDGES);
  k_finalize<<<fin_blocks, T, 0, stream>>>(buf1, h16, buf2, dinv, b1, N_NODES);

  // Layer 2 -> buf2 (f32)
  k_gemm_node<<<gemm_blocks, T, 0, stream>>>(h16, w2s, buf2);
  k_zero<<<4096, T, 0, stream>>>(buf1, NH);
  k_aggregate<<<agg_blocks, T, 0, stream>>>(buf2, src, dst, dinv, buf1, N_EDGES);
  k_finalize<<<fin_blocks, T, 0, stream>>>(buf2, h16, buf1, dinv, b2, N_NODES);

  // Mean pool
  k_zero<<<33, T, 0, stream>>>(sums, N_GRAPHS * DIM + N_GRAPHS);
  k_pool<<<(int)(((long)N_NODES * 32 + T - 1) / T), T, 0, stream>>>(buf2, batch, sums, cnts, N_NODES);
  k_pool_div<<<(N_GRAPHS * DIM + T - 1) / T, T, 0, stream>>>(sums, cnts);

  // MLP head + softmax
  k_head<<<1, T, 0, stream>>>(sums, fW1, fb1, fW2, fb2, out);
}
